// QuantLinear_14207751815307
// MI455X (gfx1250) — matrix operand order repaired and checked by reading, not compiled, not run on hardware
//
#include <hip/hip_runtime.h>

typedef __attribute__((ext_vector_type(16))) __bf16 v16bf;
typedef __attribute__((ext_vector_type(8)))  __bf16 v8bf;
typedef __attribute__((ext_vector_type(4)))  __bf16 v4bf;
typedef __attribute__((ext_vector_type(2)))  __bf16 v2bf;
typedef __attribute__((ext_vector_type(8)))  float  v8f;
typedef __attribute__((ext_vector_type(4)))  float  v4f;
typedef __attribute__((ext_vector_type(4)))  int    v4i;

#define M_TOT 16384   // B*S = 4*4096
#define N_TOT 4096    // OC
#define K_TOT 4096    // IC
#define BM 128
#define BN 128
#define BK 64
#define LDS_PITCH 72  // bf16/row: 64 + 8 pad -> 144B rows, breaks bank-stride pattern

__global__ __launch_bounds__(256) void qlinear_wmma_bf16(
    const float* __restrict__ inp,      // [M_TOT, K_TOT] f32
    const int*   __restrict__ qweight,  // [N_TOT, K_TOT/2] one byte (2 nibbles) per int32
    const float* __restrict__ scale,    // [N_TOT]
    const float* __restrict__ zero,     // [N_TOT]
    const float* __restrict__ bias,     // [N_TOT]
    float* __restrict__ out)            // [M_TOT, N_TOT] f32
{
    __shared__ __bf16 lA[2][BM * LDS_PITCH];
    __shared__ __bf16 lB[2][BN * LDS_PITCH];

    const int tid  = threadIdx.x;
    const int lane = tid & 31;
    const int wave = tid >> 5;

    const int m0 = blockIdx.y * BM;
    const int n0 = blockIdx.x * BN;

    // 8 waves: 4 along M (32 rows each), 2 along N (64 cols each)
    const int wm = (wave & 3) * 32;
    const int wn = (wave >> 2) * 64;

    // staging assignment: 2 threads per tile row, each covers 32 K-elements
    const int rowT  = tid >> 1;
    const int halfT = tid & 1;

    // per-thread dequant constants (each thread stages exactly one weight row)
    const float sc  = scale[n0 + rowT];
    const float nzs = -zero[n0 + rowT] * sc;   // w = fma(nibble, sc, nzs)

    const float* gA = inp + (size_t)(m0 + rowT) * K_TOT + halfT * 32;
    const int*   gQ = qweight + (size_t)(n0 + rowT) * (K_TOT / 2) + halfT * 16;

    v8f acc[2][4];
#pragma unroll
    for (int i = 0; i < 2; ++i)
#pragma unroll
        for (int j = 0; j < 4; ++j)
#pragma unroll
            for (int e = 0; e < 8; ++e) acc[i][j][e] = 0.0f;

    // register staging for the in-flight tile
    v4f aReg[8];
    v4i qReg[4];

    auto loadG = [&](int k0) {
#pragma unroll
        for (int v = 0; v < 8; ++v) aReg[v] = *(const v4f*)(gA + k0 + v * 4);
#pragma unroll
        for (int v = 0; v < 4; ++v) qReg[v] = *(const v4i*)(gQ + (k0 >> 1) + v * 4);
    };

    auto stageLDS = [&](int buf) {
        __bf16* lpA = &lA[buf][rowT * LDS_PITCH + halfT * 32];
#pragma unroll
        for (int v = 0; v < 8; ++v) {
            v4bf h;
            h[0] = (__bf16)aReg[v][0]; h[1] = (__bf16)aReg[v][1];
            h[2] = (__bf16)aReg[v][2]; h[3] = (__bf16)aReg[v][3];
            *(v4bf*)(lpA + v * 4) = h;
        }
        __bf16* lpB = &lB[buf][rowT * LDS_PITCH + halfT * 32];
#pragma unroll
        for (int v = 0; v < 4; ++v) {
#pragma unroll
            for (int e = 0; e < 4; ++e) {
                const int b = qReg[v][e];                 // byte value 0..255
                v2bf w;
                w[0] = (__bf16)__builtin_fmaf((float)(b >> 4), sc, nzs);  // even col
                w[1] = (__bf16)__builtin_fmaf((float)(b & 15), sc, nzs);  // odd col
                *(v2bf*)(lpB + (v * 4 + e) * 2) = w;
            }
        }
    };

    const int h   = lane >> 4;   // lane half selects K sub-chunk / M+8
    const int l16 = lane & 15;

    auto compute = [&](int buf) {
#pragma unroll
        for (int kk = 0; kk < BK; kk += 32) {
            // A fragments: 16x32 bf16; lane row = l16, chunks K = kk+h*8 and kk+16+h*8
            v16bf afrag[2];
#pragma unroll
            for (int i = 0; i < 2; ++i) {
                const __bf16* base = &lA[buf][(wm + i * 16 + l16) * LDS_PITCH + kk + h * 8];
                v8bf lo8 = *(const v8bf*)(base);
                v8bf hi8 = *(const v8bf*)(base + 16);
                afrag[i] = __builtin_shufflevector(lo8, hi8,
                    0,1,2,3,4,5,6,7, 8,9,10,11,12,13,14,15);
            }
            // B fragments: 32x16 bf16; lane col = l16, chunks K = kk+h*8 and kk+16+h*8
            v16bf bfrag[4];
#pragma unroll
            for (int j = 0; j < 4; ++j) {
                const __bf16* base = &lB[buf][(wn + j * 16 + l16) * LDS_PITCH + kk + h * 8];
                v8bf lo8 = *(const v8bf*)(base);
                v8bf hi8 = *(const v8bf*)(base + 16);
                bfrag[j] = __builtin_shufflevector(lo8, hi8,
                    0,1,2,3,4,5,6,7, 8,9,10,11,12,13,14,15);
            }
#pragma unroll
            for (int i = 0; i < 2; ++i)
#pragma unroll
                for (int j = 0; j < 4; ++j)
                    acc[i][j] = __builtin_amdgcn_wmma_f32_16x16x32_bf16(
                        false, afrag[i], false, bfrag[j],
                        (short)0, acc[i][j], false, false);
        }
    };

    // ---- software pipeline: double-buffered LDS, register-prefetched globals ----
    loadG(0);
    stageLDS(0);
    int buf = 0;
    for (int k0 = 0; k0 < K_TOT; k0 += BK) {
        const bool hasNext = (k0 + BK) < K_TOT;
        if (hasNext) loadG(k0 + BK);              // issue early: latency hidden by WMMAs
        if (k0 + 2 * BK < K_TOT) {                // L2 prefetch at distance 2
            __builtin_prefetch(gA + k0 + 2 * BK, 0, 0);
            __builtin_prefetch(gQ + ((k0 + 2 * BK) >> 1), 0, 0);
        }
        __syncthreads();                          // lds[buf] ready; lds[buf^1] drained
        compute(buf);
        if (hasNext) stageLDS(buf ^ 1);
        buf ^= 1;
    }

    // ---- epilogue: bias add + store (C/D layout: lane half -> M+8, VGPR r -> M row) ----
#pragma unroll
    for (int j = 0; j < 4; ++j) {
        const int col = n0 + wn + j * 16 + l16;
        const float bv = bias[col];
#pragma unroll
        for (int i = 0; i < 2; ++i) {
            const int rowBase = m0 + wm + i * 16 + h * 8;
#pragma unroll
            for (int r = 0; r < 8; ++r) {
                out[(size_t)(rowBase + r) * N_TOT + col] = acc[i][j][r] + bv;
            }
        }
    }
}

extern "C" void kernel_launch(void* const* d_in, const int* in_sizes, int n_in,
                              void* d_out, int out_size, void* d_ws, size_t ws_size,
                              hipStream_t stream) {
    const float* inp   = (const float*)d_in[0];  // [4,4096,4096] f32
    const int*   qw    = (const int*)d_in[1];    // [4096,2048] int32 (byte values)
    const float* scale = (const float*)d_in[2];  // [4096,1]
    const float* zero  = (const float*)d_in[3];  // [4096,1]
    const float* bias  = (const float*)d_in[4];  // [4096]
    float* out = (float*)d_out;                  // [4,4096,4096] f32

    dim3 grid(N_TOT / BN, M_TOT / BM);           // 32 x 128 workgroups
    qlinear_wmma_bf16<<<grid, 256, 0, stream>>>(inp, qw, scale, zero, bias, out);
}
